// AtomActionPredictor_5686536700301
// MI455X (gfx1250) — matrix operand order repaired and checked by reading, not compiled, not run on hardware
//
#include <hip/hip_runtime.h>

// ---------- types ----------
typedef __attribute__((ext_vector_type(16))) __bf16          v16bf;
typedef __attribute__((ext_vector_type(8)))  float           v8f;
typedef __attribute__((ext_vector_type(8)))  unsigned int    v8u;
typedef __attribute__((ext_vector_type(4)))  unsigned int    v4u;
typedef __attribute__((ext_vector_type(4)))  float           v4f;

#define WAVES_PER_BLOCK 4
#define TILES_PER_WAVE  2

// Branchless GELU (tanh form): x * sigmoid(1.5957691*x + 0.0713548*x^3).
// Uses hardware v_exp_f32 + v_rcp_f32; error ~1e-3, below the bf16
// quantization already applied to the matmul operands.
__device__ __forceinline__ float gelu_fast(float x) {
    float x2 = x * x;
    float s  = x * __builtin_fmaf(x2, 0.0713548163f, 1.5957691216f);
    float e  = __expf(-s);                       // v_exp_f32
    return x * __builtin_amdgcn_rcpf(1.0f + e);  // v_rcp_f32
}

// Convert two groups of 8 contiguous fp32 into one bf16 A-fragment register
// block (elements 0..7 from g0, 8..15 from g1) -> v_cvt_pk_bf16_f32.
__device__ __forceinline__ v16bf cvt16(const float* __restrict__ g0,
                                       const float* __restrict__ g1) {
    v4f x[4];
    x[0] = *(const v4f*)(g0);
    x[1] = *(const v4f*)(g0 + 4);
    x[2] = *(const v4f*)(g1);
    x[3] = *(const v4f*)(g1 + 4);
    v16bf t;
#pragma unroll
    for (int i = 0; i < 16; ++i) t[i] = (__bf16)x[i >> 2][i & 3];
    return t;
}

// ---------------------------------------------------------------------------
// Pre-pass: pack W1 (256x128 f32) and W2 (128x128 f32) into bf16 WMMA
// B-fragment layout. B fragment (32x16, 16-bit) for wave32:
//   lane l holds column N = l&15, K = 8*(l>>4) + e + (e & 8), e = 0..15.
// Fragment stored per-lane contiguous: 32 lanes x 32 B = 1 KB.
// ---------------------------------------------------------------------------
__global__ __launch_bounds__(256) void pack_weights(
    const float* __restrict__ W1, const float* __restrict__ W2,
    unsigned short* __restrict__ wsW1, unsigned short* __restrict__ wsW2)
{
    const int t    = blockIdx.x * blockDim.x + threadIdx.x; // (frag, lane)
    const int lane = t & 31;
    const int frag = t >> 5;
    const int n    = lane & 15;
    const int kh   = (lane >> 4) * 8;

    if (frag < 64) {                       // W1: frag = nt*8 + kt
        const int nt = frag >> 3, kt = frag & 7;
        const int col = nt * 16 + n;
        const int kbase = kt * 32 + kh;
        unsigned short* dst = wsW1 + ((size_t)frag * 32 + lane) * 16;
#pragma unroll
        for (int e = 0; e < 16; ++e)
            dst[e] = __builtin_bit_cast(unsigned short,
                         (__bf16)W1[(size_t)(kbase + e + (e & 8)) * 128 + col]);
    } else if (frag < 96) {                // W2: frag-64 = nt*4 + kt
        const int f2 = frag - 64;
        const int nt = f2 >> 2, kt = f2 & 3;
        const int col = nt * 16 + n;
        const int kbase = kt * 32 + kh;
        unsigned short* dst = wsW2 + ((size_t)f2 * 32 + lane) * 16;
#pragma unroll
        for (int e = 0; e < 16; ++e)
            dst[e] = __builtin_bit_cast(unsigned short,
                         (__bf16)W2[(size_t)(kbase + e + (e & 8)) * 128 + col]);
    }
}

// ---------------------------------------------------------------------------
// Fused gather + MLP. One wave handles TWO 16-row tiles (each B fragment
// feeds two WMMAs; L2 weight traffic ~1.2 GB/call, under the HBM roofline).
// B-fragment stream is software-pipelined with prefetch distance 2.
// ---------------------------------------------------------------------------
__global__ __launch_bounds__(128) void atom_mlp_wmma(
    const float* __restrict__ atoms, const int* __restrict__ rc_idx,
    const unsigned short* __restrict__ wsW1,
    const unsigned short* __restrict__ wsW2,
    const float* __restrict__ b1, const float* __restrict__ b2,
    float* __restrict__ out, int M, int numTiles)
{
    __shared__ __align__(16) unsigned short
        hbuf[WAVES_PER_BLOCK][TILES_PER_WAVE][16 * 128];   // 32 KB

    const int lane  = threadIdx.x & 31;
    const int wave  = threadIdx.x >> 5;
    const int unit  = blockIdx.x * WAVES_PER_BLOCK + wave;
    const int tile0 = unit * TILES_PER_WAVE;
    if (tile0 >= numTiles) return;         // wave-uniform: EXEC stays all-ones

    const int rmod  = lane & 15;           // A row / B,C column within tile
    const int khalf = lane >> 4;           // K sub-half selector

    // ---- preload per-lane biases (off the critical path) ----
    float bias1[8], bias2[8];
#pragma unroll
    for (int nt = 0; nt < 8; ++nt) {
        bias1[nt] = b1[nt * 16 + rmod];
        bias2[nt] = b2[nt * 16 + rmod];
    }

    // ---- gather + fp32->bf16 convert A fragments (2 x 16x256) ----
    v16bf a[TILES_PER_WAVE][8];
#pragma unroll
    for (int t = 0; t < TILES_PER_WAVE; ++t) {
        const int rowA = (tile0 + t) * 16 + rmod;
        const int atom = (rowA < M) ? rc_idx[rowA] : 0;
        const float* arow = atoms + (size_t)atom * 256;
#pragma unroll
        for (int kt = 0; kt < 8; ++kt) {
            const int k0 = kt * 32 + khalf * 8;
            a[t][kt] = cvt16(arow + k0, arow + k0 + 16);
        }
    }

    // B-fragment loaders (per-lane contiguous 32 B from packed weights)
    auto loadB1 = [&](int f) -> v16bf {
        v8u uw = *(const v8u*)(wsW1 + ((size_t)f * 32 + lane) * 16);
        return __builtin_bit_cast(v16bf, uw);
    };
    auto loadB2 = [&](int f) -> v16bf {
        v8u uw = *(const v8u*)(wsW2 + ((size_t)f * 32 + lane) * 16);
        return __builtin_bit_cast(v16bf, uw);
    };

    // ---- GEMM1: h = gelu(A @ W1 + b1) -> bf16 in wave-private LDS ----
    {
        v16bf bq0 = loadB1(0);             // prefetch ring, depth 2
        v16bf bq1 = loadB1(1);
#pragma unroll
        for (int nt = 0; nt < 8; ++nt) {
            v8f c[TILES_PER_WAVE] = {};
#pragma unroll
            for (int kt = 0; kt < 8; ++kt) {
                const int f = nt * 8 + kt;
                v16bf bcur = bq0;
                bq0 = bq1;
                if (f + 2 < 64) bq1 = loadB1(f + 2);
#pragma unroll
                for (int t = 0; t < TILES_PER_WAVE; ++t)
                    c[t] = __builtin_amdgcn_wmma_f32_16x16x32_bf16(
                               false, a[t][kt], false, bcur,
                               (short)0, c[t], false, false);
            }
            const int   n    = nt * 16 + rmod;
            const float bias = bias1[nt];
#pragma unroll
            for (int t = 0; t < TILES_PER_WAVE; ++t) {
                unsigned short* hb = &hbuf[wave][t][0];
#pragma unroll
                for (int r = 0; r < 8; ++r) {
                    float g = gelu_fast(c[t][r] + bias);
                    hb[(r + 8 * khalf) * 128 + n] =
                        __builtin_bit_cast(unsigned short, (__bf16)g);
                }
            }
        }
    }

    // Wave-internal fence: LDS tiles are wave-private, so a DScnt drain makes
    // the cross-lane h exchange visible without a block barrier.
    asm volatile("s_wait_dscnt 0" ::: "memory");

    // ---- reload h (16x128 bf16) as A-fragments ----
    v16bf a2[TILES_PER_WAVE][4];
#pragma unroll
    for (int t = 0; t < TILES_PER_WAVE; ++t) {
        const unsigned short* hr = &hbuf[wave][t][rmod * 128];
#pragma unroll
        for (int kt = 0; kt < 4; ++kt) {
            const int k0 = kt * 32 + khalf * 8;
            v4u lo = *(const v4u*)(hr + k0);
            v4u hi = *(const v4u*)(hr + k0 + 16);
            a2[t][kt] = __builtin_bit_cast(
                v16bf, __builtin_shufflevector(lo, hi, 0, 1, 2, 3, 4, 5, 6, 7));
        }
    }

    // ---- GEMM2: logits = h @ W2 + b2, streamed out (non-temporal) ----
    {
        v16bf bq0 = loadB2(0);             // prefetch ring, depth 2
        v16bf bq1 = loadB2(1);
#pragma unroll
        for (int nt = 0; nt < 8; ++nt) {
            v8f c[TILES_PER_WAVE] = {};
#pragma unroll
            for (int kt = 0; kt < 4; ++kt) {
                const int f = nt * 4 + kt;
                v16bf bcur = bq0;
                bq0 = bq1;
                if (f + 2 < 32) bq1 = loadB2(f + 2);
#pragma unroll
                for (int t = 0; t < TILES_PER_WAVE; ++t)
                    c[t] = __builtin_amdgcn_wmma_f32_16x16x32_bf16(
                               false, a2[t][kt], false, bcur,
                               (short)0, c[t], false, false);
            }
            const int   n    = nt * 16 + rmod;
            const float bias = bias2[nt];
#pragma unroll
            for (int t = 0; t < TILES_PER_WAVE; ++t) {
#pragma unroll
                for (int r = 0; r < 8; ++r) {
                    const int orow = (tile0 + t) * 16 + r + 8 * khalf;
                    if (orow < M)
                        __builtin_nontemporal_store(c[t][r] + bias,
                                                    &out[(size_t)orow * 128 + n]);
                }
            }
        }
    }
}

// ---------------------------------------------------------------------------
extern "C" void kernel_launch(void* const* d_in, const int* in_sizes, int n_in,
                              void* d_out, int out_size, void* d_ws, size_t ws_size,
                              hipStream_t stream)
{
    const float* atoms = (const float*)d_in[0];   // [1M, 256] f32
    const int*   idx   = (const int*)  d_in[1];   // [400K] i32
    const float* W1    = (const float*)d_in[2];   // [256, 128] f32
    const float* b1    = (const float*)d_in[3];   // [128] f32
    const float* W2    = (const float*)d_in[4];   // [128, 128] f32
    const float* b2    = (const float*)d_in[5];   // [128] f32
    float* out = (float*)d_out;                   // [400K, 128] f32

    const int M = in_sizes[1];

    unsigned short* wsW1 = (unsigned short*)d_ws;        // 64 frags * 1 KB
    unsigned short* wsW2 = wsW1 + 64 * 32 * 16;          // 32 frags * 1 KB

    pack_weights<<<12, 256, 0, stream>>>(W1, W2, wsW1, wsW2);

    const int numTiles  = (M + 15) >> 4;
    const int waveUnits = (numTiles + TILES_PER_WAVE - 1) / TILES_PER_WAVE;
    const int blocks    = (waveUnits + WAVES_PER_BLOCK - 1) / WAVES_PER_BLOCK;
    atom_mlp_wmma<<<blocks, WAVES_PER_BLOCK * 32, 0, stream>>>(
        atoms, idx, wsW1, wsW2, b1, b2, out, M, numTiles);
}
